// VectorQuantize_37323265802747
// MI455X (gfx1250) — matrix operand order repaired and checked by reading, not compiled, not run on hardware
//
#include <hip/hip_runtime.h>
#include <hip/hip_bf16.h>

typedef __attribute__((ext_vector_type(16))) _Float16 v16h;
typedef __attribute__((ext_vector_type(4)))  _Float16 v4h;
typedef __attribute__((ext_vector_type(8)))  float    v8f;

#define N_TOKENS 32768
#define N_CODES  4096
#define EMB      256
#define COMMIT   0.25f

#define N_PARTS     4
#define PART_CODES  (N_CODES / N_PARTS)          // 1024 codes per partition
#define STAGE_CODES 64
#define ROW_STRIDE  528                          // 512B row + 16B pad -> conflict-free
#define STAGE_BYTES (STAGE_CODES * ROW_STRIDE)   // 33792
#define NSTAGES_P   (PART_CODES / STAGE_CODES)   // 16 stages per partition

// ---------------- fragment loaders (contiguous k = 16*half + s, same order for A and B) ----------
// A 16x32 f16: lanes 0-15 hold K=0..15, lanes 16-31 K=16..31, M = lane&15
__device__ __forceinline__ v16h load_a_frag(const _Float16* xrow, int kbase, int half_id) {
    v16h a;
    ((float4*)&a)[0] = *(const float4*)(xrow + kbase + half_id * 16);
    ((float4*)&a)[1] = *(const float4*)(xrow + kbase + 8 + half_id * 16);
    return a;
}
// 32 contiguous bytes from LDS (B 32x16 f16: lanes 0-15 K=0..15, lanes 16-31 K=16..31)
__device__ __forceinline__ void load16x2(v16h& b, const char* p) {
    ((float4*)&b)[0] = ((const float4*)p)[0];
    ((float4*)&b)[1] = ((const float4*)p)[1];
}

// ---------------- CDNA5 async global->LDS copy helpers ----------------------
__device__ __forceinline__ void async_copy16(unsigned int lds_addr, unsigned long long gaddr) {
    asm volatile("global_load_async_to_lds_b128 %0, %1, off"
                 :: "v"(lds_addr), "v"(gaddr) : "memory");
}
__device__ __forceinline__ void wait_async8() {
    asm volatile("s_wait_asynccnt 0x8" ::: "memory");
}
__device__ __forceinline__ void wait_async0() {
    asm volatile("s_wait_asynccnt 0x0" ::: "memory");
}

// stage = 64 codebook rows (32KB), 2048 x 16B chunks, 8 per thread, padded rows
__device__ __forceinline__ void issue_stage(unsigned int lds_base, const _Float16* src, int tid) {
    unsigned long long g0 = (unsigned long long)(uintptr_t)src;
    #pragma unroll
    for (int j = 0; j < 8; ++j) {
        int c   = j * 256 + tid;               // chunk id [0,2048)
        int r   = c >> 5;                      // row (32 chunks of 16B per 512B row)
        int col = (c & 31) * 16;
        async_copy16(lds_base + r * ROW_STRIDE + col, g0 + (unsigned long long)c * 16u);
    }
}

// ---------------- kernel 1: convert x f32 -> f16 ---------------------------
__global__ void vq_cvt_x(const float* __restrict__ in, _Float16* __restrict__ out) {
    int i = blockIdx.x * blockDim.x + threadIdx.x;   // groups of 4
    float4 v = ((const float4*)in)[i];
    v4h h = { (_Float16)v.x, (_Float16)v.y, (_Float16)v.z, (_Float16)v.w };
    ((v4h*)out)[i] = h;
}

// ---------------- kernel 2: codebook convert + 0.5*||e||^2 -----------------
__global__ void vq_cb_prep(const float* __restrict__ cb, _Float16* __restrict__ cbh,
                           float* __restrict__ hnorm) {
    int r = blockIdx.x, d = threadIdx.x;
    float v = cb[(size_t)r * EMB + d];
    cbh[(size_t)r * EMB + d] = (_Float16)v;
    float s = v * v;
    for (int off = 16; off; off >>= 1) s += __shfl_xor(s, off, 32);
    __shared__ float part[8];
    if ((threadIdx.x & 31) == 0) part[threadIdx.x >> 5] = s;
    __syncthreads();
    if (threadIdx.x == 0) {
        float t = 0.f;
        #pragma unroll
        for (int w = 0; w < 8; ++w) t += part[w];
        hnorm[r] = 0.5f * t;
    }
}

// ---------------- kernel 3: WMMA distance + partial argmin ------------------
// grid (token_groups=128, parts=4). Each block: 8 waves x 32 tokens, sweeps
// PART_CODES=1024 codes staged through LDS (double-buffered async copies).
__global__ __launch_bounds__(256) void vq_argmin(const _Float16* __restrict__ xh,
                                                 const _Float16* __restrict__ cbh,
                                                 const float* __restrict__ hnorm,
                                                 float* __restrict__ cand_val,
                                                 int* __restrict__ cand_idx) {
    __shared__ char smem[2 * STAGE_BYTES + PART_CODES * 4];  // 2 stage bufs + hnorm slice
    float* s_hn = (float*)(smem + 2 * STAGE_BYTES);

    const int tid     = threadIdx.x;
    const int lane    = tid & 31;
    const int wave    = tid >> 5;
    const int l       = lane & 15;
    const int half_id = lane >> 4;
    const int row0    = (blockIdx.x * 8 + wave) * 32;
    const int part    = blockIdx.y;
    const int cbase   = part * PART_CODES;

    const _Float16* cbh_p = cbh + (size_t)cbase * EMB;

    // preload this partition's 0.5*||e||^2 slice into LDS
    for (int i = tid; i < PART_CODES; i += 256) s_hn[i] = hnorm[cbase + i];

    // resident A fragments: 32 tokens per wave
    v16h a0[8], a1[8];
    const _Float16* xr0 = xh + (size_t)(row0 + l) * EMB;
    const _Float16* xr1 = xh + (size_t)(row0 + 16 + l) * EMB;
    #pragma unroll
    for (int c = 0; c < 8; ++c) {
        a0[c] = load_a_frag(xr0, 32 * c, half_id);
        a1[c] = load_a_frag(xr1, 32 * c, half_id);
    }

    float minv0[8], minv1[8];
    int   mini0[8], mini1[8];
    #pragma unroll
    for (int i = 0; i < 8; ++i) {
        minv0[i] = minv1[i] = 3.4e38f;
        mini0[i] = mini1[i] = 0;
    }

    const unsigned int lds0 = (unsigned int)(uintptr_t)&smem[0];
    issue_stage(lds0, cbh_p, tid);                     // prologue: stage 0 -> buf0

    for (int s = 0; s < NSTAGES_P; ++s) {
        const int buf = s & 1;
        if (s + 1 < NSTAGES_P) {
            issue_stage(lds0 + (unsigned)((s + 1) & 1) * STAGE_BYTES,
                        cbh_p + (size_t)(s + 1) * STAGE_CODES * EMB, tid);
            wait_async8();                             // our stage-s copies complete
        } else {
            wait_async0();
        }
        __syncthreads();                               // all waves' stage-s data in LDS

        const char* sbuf = smem + (size_t)buf * STAGE_BYTES;
        #pragma unroll
        for (int t = 0; t < 4; ++t) {                  // 4 N-tiles of 16 codes
            const int lcol0 = s * STAGE_CODES + t * 16;     // partition-local col base
            const char* brow = sbuf + (size_t)(t * 16 + l) * ROW_STRIDE + 32 * half_id;
            const float hn = s_hn[lcol0 + l];

            // software-pipelined B fragments (2 chunks ahead) hide ds latency
            v16h b0, b1;
            load16x2(b0, brow + 0);
            load16x2(b1, brow + 64);
            v8f acc0 = {}, acc1 = {};
            #pragma unroll
            for (int c = 0; c < 8; ++c) {
                v16h bn;
                if (c < 6) load16x2(bn, brow + 64 * (c + 2));
                acc0 = __builtin_amdgcn_wmma_f32_16x16x32_f16(false, a0[c], false, b0,
                                                              (short)0, acc0, false, false);
                acc1 = __builtin_amdgcn_wmma_f32_16x16x32_f16(false, a1[c], false, b0,
                                                              (short)0, acc1, false, false);
                b0 = b1;
                if (c < 6) b1 = bn;
            }

            const int col = cbase + lcol0 + l;         // global code index
            #pragma unroll
            for (int i = 0; i < 8; ++i) {
                float d0 = hn - acc0[i];               // 0.5||e||^2 - x.e  (argmin proxy)
                if (d0 < minv0[i]) { minv0[i] = d0; mini0[i] = col; }
                float d1 = hn - acc1[i];
                if (d1 < minv1[i]) { minv1[i] = d1; mini1[i] = col; }
            }
        }
        __syncthreads();                               // buf safe to overwrite next iter
    }

    // argmin across the 16 lanes of each half-wave (C layout: M = i + 8*half_id, N = lane&15)
    #pragma unroll
    for (int i = 0; i < 8; ++i) {
        #pragma unroll
        for (int off = 8; off; off >>= 1) {
            float ov = __shfl_xor(minv0[i], off, 32);
            int   oi = __shfl_xor(mini0[i], off, 32);
            if (ov < minv0[i] || (ov == minv0[i] && oi < mini0[i])) { minv0[i] = ov; mini0[i] = oi; }
            float ow = __shfl_xor(minv1[i], off, 32);
            int   oj = __shfl_xor(mini1[i], off, 32);
            if (ow < minv1[i] || (ow == minv1[i] && oj < mini1[i])) { minv1[i] = ow; mini1[i] = oj; }
        }
        if (l == 0) {
            size_t o0 = (size_t)part * N_TOKENS + row0 + half_id * 8 + i;
            size_t o1 = o0 + 16;
            cand_val[o0] = minv0[i];  cand_idx[o0] = mini0[i];
            cand_val[o1] = minv1[i];  cand_idx[o1] = mini1[i];
        }
    }
}

// ---------------- kernel 4: merge partition candidates ----------------------
__global__ void vq_merge(const float* __restrict__ cval, const int* __restrict__ cidx,
                         int* __restrict__ idx_out) {
    int t = blockIdx.x * blockDim.x + threadIdx.x;
    float best = cval[t];
    int   bi   = cidx[t];
    #pragma unroll
    for (int p = 1; p < N_PARTS; ++p) {
        float v = cval[(size_t)p * N_TOKENS + t];
        int   j = cidx[(size_t)p * N_TOKENS + t];
        if (v < best || (v == best && j < bi)) { best = v; bi = j; }
    }
    idx_out[t] = bi;
}

// ---------------- kernel 5: gather latent + per-token loss partial ----------
__global__ void vq_gather(const float* __restrict__ x, const float* __restrict__ cb,
                          const int* __restrict__ idx, float* __restrict__ out,
                          float* __restrict__ partials) {
    int t = blockIdx.x, d = threadIdx.x;
    int k = idx[t];
    float e  = cb[(size_t)k * EMB + d];
    float xv = x[(size_t)t * EMB + d];
    out[1 + (size_t)t * EMB + d] = e;                  // latent = x + sg(e-x) = e
    float diff = e - xv;
    float s = diff * diff;
    for (int off = 16; off; off >>= 1) s += __shfl_xor(s, off, 32);
    __shared__ float part[8];
    if ((threadIdx.x & 31) == 0) part[threadIdx.x >> 5] = s;
    __syncthreads();
    if (threadIdx.x == 0) {
        float tt = 0.f;
        #pragma unroll
        for (int w = 0; w < 8; ++w) tt += part[w];
        partials[t] = tt;
    }
}

// ---------------- kernel 6: deterministic loss reduction --------------------
__global__ void vq_finalize(const float* __restrict__ partials, float* __restrict__ out) {
    float s = 0.f;
    for (int i = threadIdx.x; i < N_TOKENS; i += 256) s += partials[i];
    for (int off = 16; off; off >>= 1) s += __shfl_xor(s, off, 32);
    __shared__ float part[8];
    if ((threadIdx.x & 31) == 0) part[threadIdx.x >> 5] = s;
    __syncthreads();
    if (threadIdx.x == 0) {
        float t = 0.f;
        #pragma unroll
        for (int w = 0; w < 8; ++w) t += part[w];
        // loss = COMMIT * e_loss + q_loss, both numerically mean((e-x)^2)
        out[0] = (1.0f + COMMIT) * t / (float)((size_t)N_TOKENS * EMB);
    }
}

extern "C" void kernel_launch(void* const* d_in, const int* in_sizes, int n_in,
                              void* d_out, int out_size, void* d_ws, size_t ws_size,
                              hipStream_t stream) {
    const float* x  = (const float*)d_in[0];   // [32768,256] f32
    const float* cb = (const float*)d_in[1];   // [4096,256]  f32
    float* out = (float*)d_out;                // [1 + 32768*256] f32

    char* ws = (char*)d_ws;
    _Float16* xh    = (_Float16*)ws;                                    // 16 MB
    _Float16* cbh   = (_Float16*)(ws + (size_t)N_TOKENS * EMB * 2);     //  2 MB
    char* p = ws + (size_t)N_TOKENS * EMB * 2 + (size_t)N_CODES * EMB * 2;
    float* hnorm = (float*)p;            p += (size_t)N_CODES * 4;      // 16 KB
    int*   idx   = (int*)p;              p += (size_t)N_TOKENS * 4;     // 128 KB
    float* parts = (float*)p;            p += (size_t)N_TOKENS * 4;     // 128 KB
    float* cval  = (float*)p;            p += (size_t)N_PARTS * N_TOKENS * 4;  // 512 KB
    int*   cidx  = (int*)p;                                             // 512 KB

    vq_cvt_x<<<(N_TOKENS * EMB / 4) / 256, 256, 0, stream>>>(x, xh);
    vq_cb_prep<<<N_CODES, 256, 0, stream>>>(cb, cbh, hnorm);
    vq_argmin<<<dim3(N_TOKENS / (8 * 32), N_PARTS), 256, 0, stream>>>(xh, cbh, hnorm, cval, cidx);
    vq_merge<<<N_TOKENS / 256, 256, 0, stream>>>(cval, cidx, idx);
    vq_gather<<<N_TOKENS, 256, 0, stream>>>(x, cb, idx, out, parts);
    vq_finalize<<<1, 256, 0, stream>>>(parts, out);
}
